// MultiHeadAttention_86560770884100
// MI455X (gfx1250) — matrix operand order repaired and checked by reading, not compiled, not run on hardware
//
#include <hip/hip_runtime.h>
#include <hip/hip_bf16.h>

typedef __attribute__((ext_vector_type(16))) __bf16 v16bf;
typedef __attribute__((ext_vector_type(8)))  __bf16 v8bf;
typedef __attribute__((ext_vector_type(8)))  float  v8f;
typedef __attribute__((ext_vector_type(4)))  float  v4f;

#if __has_builtin(__builtin_amdgcn_tensor_load_to_lds)
#define HAVE_TDM 1
typedef __attribute__((ext_vector_type(4))) unsigned int v4u;
typedef __attribute__((ext_vector_type(8))) int v8i;
typedef __attribute__((ext_vector_type(4))) int v4i;
#else
#define HAVE_TDM 0
#endif

#define N_B  4
#define T_S  2048
#define DMODEL 1024
#define H_N  16
#define D_H  64
#define HD   1024

union V16U { v16bf v; v8bf h[2]; };

static __device__ __forceinline__ v16bf ld16(const __bf16* p0, const __bf16* p1) {
  V16U u;
  u.h[0] = *(const v8bf*)p0;
  u.h[1] = *(const v8bf*)p1;
  return u.v;
}

static __device__ __forceinline__ v8f wmma_bf16(v16bf a, v16bf b, v8f c) {
  return __builtin_amdgcn_wmma_f32_16x16x32_bf16(false, a, false, b, (short)0, c,
                                                 false, false);
}

static __device__ __forceinline__ float xmax16(float v) {
  v = fmaxf(v, __shfl_xor(v, 1, 32));
  v = fmaxf(v, __shfl_xor(v, 2, 32));
  v = fmaxf(v, __shfl_xor(v, 4, 32));
  v = fmaxf(v, __shfl_xor(v, 8, 32));
  return v;
}
static __device__ __forceinline__ float xsum16(float v) {
  v += __shfl_xor(v, 1, 32);
  v += __shfl_xor(v, 2, 32);
  v += __shfl_xor(v, 4, 32);
  v += __shfl_xor(v, 8, 32);
  return v;
}

// ---------------------------------------------------------------------------
// Tiled GEMM: C[8192,1024] = A[8192,1024] @ W[1024,1024] (+bias)
// Double-buffered LDS (one barrier per K-step); bf16 WMMA, f32 accumulate.
//   A_F32:    A is f32 (converted to bf16 during staging) else bf16.
//   HEAD_OUT: bf16 into head-major [n*H+h][t][d], scaled; else f32 row-major.
// Block 128x128, BK=32, 256 threads = 8 waves (2x4 of 64x32 sub-tiles).
// ---------------------------------------------------------------------------
template <bool A_F32, bool HEAD_OUT>
__global__ __launch_bounds__(256)
void gemm_kernel(const void* __restrict__ Aptr, const float* __restrict__ W,
                 const float* __restrict__ bias, void* __restrict__ Cptr,
                 float scale) {
  constexpr int BM = 128, BN = 128, BK = 32;
  constexpr int AST = BK + 8;  // 40: 80B rows -> 20-bank shift, conflict-free
  constexpr int BST = BK + 8;
  constexpr int ASZ = BM * AST;
  constexpr int BSZ = BN * BST;
  constexpr int NKT = DMODEL / BK;  // 32 K-steps
  __shared__ __align__(16) __bf16 Al[2 * ASZ];
  __shared__ __align__(16) __bf16 Bl[2 * BSZ];

  const int tid = threadIdx.x;
  const int lane = tid & 31;
  const int wid = tid >> 5;
  const int ln = lane & 15;
  const int hf = lane >> 4;
  const int row0 = blockIdx.y * BM;
  const int col0 = blockIdx.x * BN;
  const int wm = (wid >> 2) * 64;  // wave m-offset: 0 / 64
  const int wn = (wid & 3) * 32;   // wave n-offset: 0..96

  v8f acc[4][2];
#pragma unroll
  for (int mt = 0; mt < 4; ++mt)
#pragma unroll
    for (int nt = 0; nt < 2; ++nt)
#pragma unroll
      for (int r = 0; r < 8; ++r) acc[mt][nt][r] = 0.0f;

  // register staging for the in-flight tile
  v4f af32[4];
  v8bf ab16[2];
  v4f bw[4];

  auto load_tiles = [&](int k0) {
    if constexpr (A_F32) {
      const float* A = (const float*)Aptr;
#pragma unroll
      for (int i = 0; i < 4; ++i) {
        int c = tid + 256 * i;  // 1024 chunks of 4 f32
        int r = c >> 3;
        int cc = (c & 7) * 4;
        af32[i] = *(const v4f*)(A + (size_t)(row0 + r) * DMODEL + k0 + cc);
      }
    } else {
      const __bf16* A = (const __bf16*)Aptr;
#pragma unroll
      for (int i = 0; i < 2; ++i) {
        int c = tid + 256 * i;  // 512 chunks of 8 bf16
        int r = c >> 2;
        int cc = (c & 3) * 8;
        ab16[i] = *(const v8bf*)(A + (size_t)(row0 + r) * DMODEL + k0 + cc);
      }
    }
#pragma unroll
    for (int i = 0; i < 4; ++i) {
      int c = tid + 256 * i;  // 1024 chunks of 4 f32
      int kk = c >> 5;
      int nn = (c & 31) * 4;
      bw[i] = *(const v4f*)(W + (size_t)(k0 + kk) * HD + col0 + nn);
    }
  };

  auto store_tiles = [&](int p) {
    __bf16* Ab = &Al[p * ASZ];
    __bf16* Bb = &Bl[p * BSZ];
    if constexpr (A_F32) {
#pragma unroll
      for (int i = 0; i < 4; ++i) {
        int c = tid + 256 * i;
        int r = c >> 3;
        int cc = (c & 7) * 4;
        __bf16* dst = &Ab[r * AST + cc];
        dst[0] = (__bf16)af32[i].x; dst[1] = (__bf16)af32[i].y;
        dst[2] = (__bf16)af32[i].z; dst[3] = (__bf16)af32[i].w;
      }
    } else {
#pragma unroll
      for (int i = 0; i < 2; ++i) {
        int c = tid + 256 * i;
        int r = c >> 2;
        int cc = (c & 3) * 8;
        *(v8bf*)&Ab[r * AST + cc] = ab16[i];
      }
    }
#pragma unroll
    for (int i = 0; i < 4; ++i) {
      int c = tid + 256 * i;
      int kk = c >> 5;
      int nn = (c & 31) * 4;
      Bb[(nn + 0) * BST + kk] = (__bf16)bw[i].x;
      Bb[(nn + 1) * BST + kk] = (__bf16)bw[i].y;
      Bb[(nn + 2) * BST + kk] = (__bf16)bw[i].z;
      Bb[(nn + 3) * BST + kk] = (__bf16)bw[i].w;
    }
  };

  load_tiles(0);
  store_tiles(0);
  __syncthreads();

  for (int kt = 0; kt < NKT; ++kt) {
    const int p = kt & 1;
    if (kt + 1 < NKT) load_tiles((kt + 1) * BK);  // overlap with compute

    const __bf16* Ab = &Al[p * ASZ];
    const __bf16* Bb = &Bl[p * BSZ];
    v16bf af[4], bfr[2];
    const int kh = hf * 8;
    const int kh2 = hf * 16;
#pragma unroll
    for (int mt = 0; mt < 4; ++mt) {
      int row = wm + mt * 16 + ln;
      af[mt] = ld16(&Ab[row * AST + kh], &Ab[row * AST + kh + 16]);
    }
#pragma unroll
    for (int nt = 0; nt < 2; ++nt) {
      int ncol = wn + nt * 16 + ln;
      bfr[nt] = ld16(&Bb[ncol * BST + kh], &Bb[ncol * BST + kh + 16]);
    }
#pragma unroll
    for (int mt = 0; mt < 4; ++mt)
#pragma unroll
      for (int nt = 0; nt < 2; ++nt)
        acc[mt][nt] = wmma_bf16(af[mt], bfr[nt], acc[mt][nt]);

    if (kt + 1 < NKT) store_tiles(p ^ 1);  // other buffer: safe while p is read
    __syncthreads();
  }

  // ---- epilogue ----
#pragma unroll
  for (int mt = 0; mt < 4; ++mt) {
#pragma unroll
    for (int nt = 0; nt < 2; ++nt) {
      const int col = col0 + wn + nt * 16 + ln;
      const float bv = bias[col];
#pragma unroll
      for (int r = 0; r < 8; ++r) {
        const int row = row0 + wm + mt * 16 + hf * 8 + r;
        float v = acc[mt][nt][r] + bv;
        if constexpr (HEAD_OUT) {
          v *= scale;
          const int n = row >> 11;
          const int t = row & (T_S - 1);
          const int h = col >> 6;
          const int d = col & 63;
          ((__bf16*)Cptr)[((size_t)(n * H_N + h) * T_S + t) * D_H + d] = (__bf16)v;
        } else {
          ((float*)Cptr)[(size_t)row * DMODEL + col] = v;
        }
      }
    }
  }
}

// ---------------------------------------------------------------------------
// Flash attention: grid (T/128, N*H). 256 threads = 8 waves; wave w owns the
// 16-query strip [q0+16w, q0+16w+15]. Key blocks of 64 staged in LDS (K tile
// via Tensor Data Mover with pad_enable reproducing the 144B padded stride).
// ---------------------------------------------------------------------------
__global__ __launch_bounds__(256)
void flash_kernel(const __bf16* __restrict__ Qh, const __bf16* __restrict__ Kh,
                  const __bf16* __restrict__ Vh, const int* __restrict__ amask,
                  __bf16* __restrict__ Aout) {
  constexpr int BQ = 128, BS = 64;
  constexpr int KST = 72, VST = 72, PST = 72;  // padded strides (144B rows)
  __shared__ __align__(16) __bf16 Kl[BS * KST];
  __shared__ __align__(16) __bf16 Vl[D_H * VST];
  __shared__ __align__(16) __bf16 Pl[8 * 16 * PST];
  __shared__ int Ml[BS];

  const int tid = threadIdx.x;
  const int lane = tid & 31;
  const int w = tid >> 5;
  const int ln = lane & 15;
  const int hf = lane >> 4;
  const int qb = blockIdx.x;
  const int nh = blockIdx.y;
  const int n = nh >> 4;
  const int h = nh & 15;
  const int q0 = qb * BQ;
  const size_t base = (size_t)nh * T_S * D_H;

  // Q fragments for this wave's 16-row strip (held for whole kernel)
  const int tq = q0 + w * 16 + ln;
  v16bf qf[2];
#pragma unroll
  for (int ks = 0; ks < 2; ++ks) {
    const __bf16* qp = Qh + base + (size_t)tq * D_H + hf * 8 + ks * 32;
    qf[ks] = ld16(qp, qp + 16);
  }

  float mx[8], ls[8];
  v8f accO[4];
#pragma unroll
  for (int r = 0; r < 8; ++r) { mx[r] = -1e30f; ls[r] = 0.0f; }
#pragma unroll
  for (int dt = 0; dt < 4; ++dt)
#pragma unroll
    for (int r = 0; r < 8; ++r) accO[dt][r] = 0.0f;

  const int kb_end = 2 * qb + 1;  // last two blocks need causal masking
  for (int kb = 0; kb <= kb_end; ++kb) {
    const int s0 = kb * BS;

#if HAVE_TDM
    // ---- K tile via Tensor Data Mover: 64 rows x 128B, LDS-padded +16B/row
    if (w == 0) {
      const unsigned long long ga =
          (unsigned long long)(const void*)(Kh + base + (size_t)s0 * D_H);
      v4u g0;
      g0[0] = 1u;                                    // count=1 (valid user D#)
      g0[1] = (unsigned int)(size_t)&Kl[0];          // lds_addr (bytes)
      g0[2] = (unsigned int)(ga & 0xffffffffu);      // global_addr[31:0]
      g0[3] = (unsigned int)((ga >> 32) & 0x1ffffffu) | (2u << 30);  // type=2
      v8i g1;
      g1[0] = (3 << 16) | (1 << 20) | (4 << 22) | (3 << 25);
      //       data=8B     pad_en     interval=32dw  amount=4dw (16B pad)
      g1[1] = (int)(16u << 16);  // tensor_dim0 = 16 (8B units = 128B row)
      g1[2] = (int)(64u << 16);  // tensor_dim1 = 64 rows
      g1[3] = (int)(16u << 16);  // tile_dim0   = 16
      g1[4] = 64;                // tile_dim1   = 64
      g1[5] = 16;                // tensor_dim0_stride = 16 (contiguous)
      g1[6] = 0;
      g1[7] = 0;
      v4i gz4 = {0, 0, 0, 0};
      v8i gz8 = {0, 0, 0, 0, 0, 0, 0, 0};
      __builtin_amdgcn_tensor_load_to_lds(g0, g1, gz4, gz4, gz8, 0);
    }
#else
#pragma unroll
    for (int i = 0; i < 2; ++i) {
      int c = tid + 256 * i;
      int sr = c >> 3;
      int cc = (c & 7) * 8;
      v8bf d = *(const v8bf*)(Kh + base + (size_t)(s0 + sr) * D_H + cc);
      *(v8bf*)&Kl[sr * KST + cc] = d;
    }
#endif
    // ---- V tile transposed (manual: TDM cannot transpose) ----
#pragma unroll
    for (int i = 0; i < 2; ++i) {
      int c = tid + 256 * i;
      int sr = c >> 3;
      int dd = (c & 7) * 8;
      v8bf d = *(const v8bf*)(Vh + base + (size_t)(s0 + sr) * D_H + dd);
#pragma unroll
      for (int j = 0; j < 8; ++j) Vl[(dd + j) * VST + sr] = d[j];
    }
    if (tid < BS) Ml[tid] = amask[n * T_S + s0 + tid];
#if HAVE_TDM
    if (w == 0) __builtin_amdgcn_s_wait_tensorcnt(0);
#endif
    __syncthreads();

    if (s0 <= q0 + w * 16 + 15) {  // wave has at least one valid (t >= s) row
      // ---- scores: 4 tiles of 16 keys ----
      v8f S[4];
#pragma unroll
      for (int j = 0; j < 4; ++j) {
        v8f s;
#pragma unroll
        for (int r = 0; r < 8; ++r) s[r] = 0.0f;
        const int srow = j * 16 + ln;
#pragma unroll
        for (int ks = 0; ks < 2; ++ks) {
          const int kh2 = hf * 8 + ks * 32;
          v16bf kf = ld16(&Kl[srow * KST + kh2], &Kl[srow * KST + kh2 + 16]);
          s = wmma_bf16(qf[ks], kf, s);
        }
        S[j] = s;
      }
      // ---- masking: causal only on the two diagonal blocks ----
      if (kb >= 2 * qb) {
#pragma unroll
        for (int j = 0; j < 4; ++j) {
          const int sg = s0 + j * 16 + ln;
          const bool pm = (Ml[j * 16 + ln] != 0);
#pragma unroll
          for (int r = 0; r < 8; ++r) {
            const int tg = q0 + w * 16 + hf * 8 + r;
            if (!pm || sg > tg) S[j][r] = -1e30f;
          }
        }
      } else {
#pragma unroll
        for (int j = 0; j < 4; ++j) {
          if (Ml[j * 16 + ln] == 0) {
#pragma unroll
            for (int r = 0; r < 8; ++r) S[j][r] = -1e30f;
          }
        }
      }
      // ---- online softmax ----
      float corr[8], rs[8];
#pragma unroll
      for (int r = 0; r < 8; ++r) {
        float rm = fmaxf(fmaxf(S[0][r], S[1][r]), fmaxf(S[2][r], S[3][r]));
        rm = xmax16(rm);
        const float nm = fmaxf(mx[r], rm);
        corr[r] = __expf(mx[r] - nm);
        mx[r] = nm;
        rs[r] = 0.0f;
      }
#pragma unroll
      for (int j = 0; j < 4; ++j)
#pragma unroll
        for (int r = 0; r < 8; ++r) {
          const float p = __expf(S[j][r] - mx[r]);
          S[j][r] = p;
          rs[r] += p;
        }
#pragma unroll
      for (int r = 0; r < 8; ++r) ls[r] = ls[r] * corr[r] + xsum16(rs[r]);
#pragma unroll
      for (int dt = 0; dt < 4; ++dt)
#pragma unroll
        for (int r = 0; r < 8; ++r) accO[dt][r] *= corr[r];

      // ---- transpose P through this wave's LDS slab ----
      __bf16* Pw = &Pl[w * 16 * PST];
#pragma unroll
      for (int j = 0; j < 4; ++j)
#pragma unroll
        for (int r = 0; r < 8; ++r)
          Pw[(hf * 8 + r) * PST + j * 16 + ln] = (__bf16)S[j][r];
      // same-wave LDS ops are in-order; compiler inserts s_wait_dscnt for loads

      // ---- O += P . V ----
#pragma unroll
      for (int ks2 = 0; ks2 < 2; ++ks2) {
        const int kh = hf * 8 + ks2 * 32;
        v16bf pf = ld16(&Pw[ln * PST + kh], &Pw[ln * PST + kh + 16]);
#pragma unroll
        for (int dt = 0; dt < 4; ++dt) {
          const int dr = dt * 16 + ln;
          const int kh2 = hf * 8 + ks2 * 32;
          v16bf vf = ld16(&Vl[dr * VST + kh2], &Vl[dr * VST + kh2 + 16]);
          accO[dt] = wmma_bf16(pf, vf, accO[dt]);
        }
      }
    }
    __syncthreads();
  }

  // ---- normalize and store to [n][t][h*64+d] bf16 ----
#pragma unroll
  for (int r = 0; r < 8; ++r) {
    const float inv = 1.0f / ls[r];
    const int t = q0 + w * 16 + hf * 8 + r;
#pragma unroll
    for (int dt = 0; dt < 4; ++dt) {
      const int d = dt * 16 + ln;
      Aout[(size_t)(n * T_S + t) * HD + h * D_H + d] = (__bf16)(accO[dt][r] * inv);
    }
  }
}

// ---------------------------------------------------------------------------
extern "C" void kernel_launch(void* const* d_in, const int* in_sizes, int n_in,
                              void* d_out, int out_size, void* d_ws, size_t ws_size,
                              hipStream_t stream) {
  (void)in_sizes; (void)n_in; (void)out_size; (void)ws_size;
  const float* xq = (const float*)d_in[0];
  const float* xk = (const float*)d_in[1];
  const float* xv = (const float*)d_in[2];
  const float* Wq = (const float*)d_in[3];
  const float* bq = (const float*)d_in[4];
  const float* Wk = (const float*)d_in[5];
  const float* bk = (const float*)d_in[6];
  const float* Wv = (const float*)d_in[7];
  const float* bv = (const float*)d_in[8];
  const float* Wo = (const float*)d_in[9];
  const float* bo = (const float*)d_in[10];
  const int* am = (const int*)d_in[11];

  __bf16* Qh = (__bf16*)d_ws;
  const size_t per = (size_t)N_B * H_N * T_S * D_H;  // 8,388,608 elems
  __bf16* Kh = Qh + per;
  __bf16* Vh = Kh + per;
  __bf16* Ao = Vh + per;  // [N*T][HD] bf16

  dim3 blk(256);
  dim3 gg(HD / 128, (N_B * T_S) / 128);  // (8, 64)
  const float scale = 0.125f;            // 1/sqrt(64), folded into Q

  gemm_kernel<true, true><<<gg, blk, 0, stream>>>(xq, Wq, bq, Qh, scale);
  gemm_kernel<true, true><<<gg, blk, 0, stream>>>(xk, Wk, bk, Kh, 1.0f);
  gemm_kernel<true, true><<<gg, blk, 0, stream>>>(xv, Wv, bv, Vh, 1.0f);

  dim3 gf(T_S / 128, N_B * H_N);  // (16, 64)
  flash_kernel<<<gf, blk, 0, stream>>>(Qh, Kh, Vh, am, Ao);

  gemm_kernel<false, false><<<gg, blk, 0, stream>>>(Ao, Wo, bo, d_out, 1.0f);
}
